// dy_multi_gat_60696477827353
// MI455X (gfx1250) — matrix operand order repaired and checked by reading, not compiled, not run on hardware
//
#include <hip/hip_runtime.h>
#include <math.h>

// ---------------------------------------------------------------------------
// Types for CDNA5 WMMA
// ---------------------------------------------------------------------------
typedef __bf16 bf16;
typedef __attribute__((ext_vector_type(16))) __bf16 v16bf;
typedef __attribute__((ext_vector_type(8)))  __bf16 v8bf;
typedef __attribute__((ext_vector_type(2)))  __bf16 bf16x2;
typedef __attribute__((ext_vector_type(8)))  float  v8f;

// ---------------------------------------------------------------------------
// Problem constants
// ---------------------------------------------------------------------------
static constexpr int T_DIM  = 12;
static constexpr int N_NODE = 10000;
static constexpr int C_DIM  = 32;
static constexpr int E_EDGE = 320000;
static constexpr int NSUP   = 3;
static constexpr int NPAD   = 10016;          // N padded to multiple of 32
static constexpr int F_DIM  = 384;            // T*C stacked feature columns
static constexpr int EPW    = 4;              // edges per wave in scatter
static constexpr int VMAIN  = (N_NODE / 32) * 32;   // 9984: mask-free main loop extent

static_assert(N_NODE % 32 == 16, "tail-peel assumes a 16-wide valid tail block");

// ---------------------------------------------------------------------------
// Workspace layout (all chunks are multiples of 256 bytes)
// ---------------------------------------------------------------------------
static constexpr size_t OFF_P    = 0;                                  // params (attn, M, P2, a1, a2, b2, c0)
static constexpr size_t SZ_P     = 16384;
static constexpr size_t OFF_H    = OFF_P + SZ_P;                       // h = x@mlp_w+b      (T,N,C) f32
static constexpr size_t SZ_H     = (size_t)T_DIM*N_NODE*C_DIM*4;
static constexpr size_t OFF_SSRC = OFF_H + SZ_H;                       // s_src (T,N) f32
static constexpr size_t SZ_S     = (size_t)T_DIM*N_NODE*4;
static constexpr size_t OFF_SDST = OFF_SSRC + SZ_S;                    // s_dst (T,N) f32
static constexpr size_t OFF_RS   = OFF_SDST + SZ_S;                    // rowsum (NSUP,T,N) f32  (zeroed)
static constexpr size_t SZ_RS    = (size_t)NSUP*T_DIM*N_NODE*4;
static constexpr size_t OFF_G    = OFF_RS + SZ_RS;                     // gat accum (NSUP,T,N,C) f32 (zeroed)
static constexpr size_t SZ_G     = (size_t)NSUP*T_DIM*N_NODE*C_DIM*4;
static constexpr size_t OFF_X1F  = OFF_G + SZ_G;                       // x1 (T,N,C) f32
static constexpr size_t OFF_X2F  = OFF_X1F + SZ_H;                     // x2 (T,N,C) f32
static constexpr size_t OFF_XADP = OFF_X2F + SZ_H;                     // x_adp (T,N,C) f32
static constexpr size_t OFF_ZBF  = OFF_XADP + SZ_H;                    // node2vec bf16 (NPAD,32)
static constexpr size_t SZ_ZBF   = (size_t)NPAD*32*2;
static constexpr size_t OFF_XT   = OFF_ZBF + SZ_ZBF;                   // X^T bf16 (F_DIM, NPAD)
static constexpr size_t SZ_XT    = (size_t)F_DIM*NPAD*2;
static constexpr size_t OFF_X1T  = OFF_XT + SZ_XT;                     // X1^T bf16 (F_DIM, NPAD)

// params region float indices
static constexpr int P_ATTN = 0;     // 64
static constexpr int P_A1   = 64;    // 32
static constexpr int P_A2   = 96;    // 32
static constexpr int P_B2   = 128;   // 32
static constexpr int P_C0   = 160;   // 1
static constexpr int P_M    = 256;   // 32x32
static constexpr int P_P2   = 1280;  // 32x32

// ---------------------------------------------------------------------------
// Tiny helpers
// ---------------------------------------------------------------------------
__device__ __forceinline__ float wred32(float v) {
#pragma unroll
  for (int m = 16; m > 0; m >>= 1) v += __shfl_xor(v, m, 32);
  return v;
}

// relu in a single VALU op: v_med3_f32(x, 0, +BIG) == clamp(x, 0, BIG)
__device__ __forceinline__ float relu1(float x) {
  return __builtin_amdgcn_fmed3f(x, 0.f, 3.0e38f);
}

// Build a 16-bit A operand (16x32, K=32) per ISA layout:
// lanes 0-15: K = 0..7 & 16..23 ; lanes 16-31: K = 8..15 & 24..31
__device__ __forceinline__ v16bf loadA16(const bf16* __restrict__ rowp, int half) {
  const v8bf lo = *(const v8bf*)(rowp + half * 8);
  const v8bf hi = *(const v8bf*)(rowp + 16 + half * 8);
  v16bf r;
#pragma unroll
  for (int i = 0; i < 8; i++) { r[i] = lo[i]; r[8 + i] = hi[i]; }
  return r;
}

__device__ __forceinline__ v8f wmma_bf16(const v16bf& a, const v16bf& b, const v8f& c) {
  return __builtin_amdgcn_wmma_f32_16x16x32_bf16(false, a, false, b, (short)0, c, false, false);
}

// ---------------------------------------------------------------------------
// K0: fold all tiny parameter algebra once per call
//   attn = (timeoh@temb_w)@atten_pool
//   M  = (Wq Wk^T)/sqrt(192), a1 = (Wq kb)/s, a2 = (Wk qb)/s, c0 = (qb.kb)/s
//   P2 = Wv Wdense, b2 = bv Wdense
// ---------------------------------------------------------------------------
__global__ void params_kernel(const float* __restrict__ timeoh, const float* __restrict__ temb_w,
                              const float* __restrict__ pool,
                              const float* __restrict__ q_w, const float* __restrict__ q_b,
                              const float* __restrict__ k_w, const float* __restrict__ k_b,
                              const float* __restrict__ v_w, const float* __restrict__ v_b,
                              const float* __restrict__ dense_w, float* __restrict__ P) {
  __shared__ float tv[8];
  const int tid = threadIdx.x;
  if (tid < 8) {
    float s = 0.f;
    for (int k = 0; k < 43; k++) s = fmaf(timeoh[k], temb_w[k * 8 + tid], s);
    tv[tid] = s;
  }
  __syncthreads();
  if (tid < 64) {
    float s = 0.f;
#pragma unroll
    for (int d = 0; d < 8; d++) s = fmaf(tv[d], pool[d * 64 + tid], s);
    P[P_ATTN + tid] = s;
  }
  const float inv_s = rsqrtf(192.0f);
  for (int idx = tid; idx < 1024; idx += blockDim.x) {
    const int i = idx >> 5, j = idx & 31;
    float m = 0.f, p2 = 0.f;
    for (int d = 0; d < 192; d++) {
      m  = fmaf(q_w[i * 192 + d], k_w[j * 192 + d], m);
      p2 = fmaf(v_w[i * 192 + d], dense_w[d * 32 + j], p2);
    }
    P[P_M + idx]  = m * inv_s;
    P[P_P2 + idx] = p2;
  }
  if (tid < 32) {
    float s1 = 0.f, s2 = 0.f, s3 = 0.f;
    for (int d = 0; d < 192; d++) {
      s1 = fmaf(q_w[tid * 192 + d], k_b[d], s1);
      s2 = fmaf(k_w[tid * 192 + d], q_b[d], s2);
      s3 = fmaf(v_b[d], dense_w[d * 32 + tid], s3);
    }
    P[P_A1 + tid] = s1 * inv_s;
    P[P_A2 + tid] = s2 * inv_s;
    P[P_B2 + tid] = s3;
  }
  if (tid == 0) {
    float s = 0.f;
    for (int d = 0; d < 192; d++) s = fmaf(q_b[d], k_b[d], s);
    P[P_C0] = s * inv_s;
  }
}

// ---------------------------------------------------------------------------
// Zeroing (ws is poisoned each run; accumulators must start at 0)
// ---------------------------------------------------------------------------
__global__ void zero_kernel(float* __restrict__ p, size_t n) {
  size_t i = (size_t)blockIdx.x * blockDim.x + threadIdx.x;
  const size_t stride = (size_t)gridDim.x * blockDim.x;
  for (; i < n; i += stride) p[i] = 0.f;
}

__global__ void padzero_kernel(bf16* __restrict__ x1t) {
  const int id = blockIdx.x * blockDim.x + threadIdx.x;
  if (id < F_DIM * (NPAD - N_NODE)) {
    const int f = id / (NPAD - N_NODE), j = id % (NPAD - N_NODE);
    x1t[(size_t)f * NPAD + N_NODE + j] = (bf16)0.f;
  }
}

// ---------------------------------------------------------------------------
// K1: h = x@mlp_w + b ; s_src/s_dst = h . attn_{lo/hi}   (one wave per row)
// ---------------------------------------------------------------------------
__global__ __launch_bounds__(256) void hs_kernel(
    const float* __restrict__ x, const float* __restrict__ mlp_w, const float* __restrict__ mlp_b,
    const float* __restrict__ P, float* __restrict__ h,
    float* __restrict__ ssrc, float* __restrict__ sdst) {
  const int row = (blockIdx.x * blockDim.x + threadIdx.x) >> 5;
  const int lane = threadIdx.x & 31;
  if (row >= T_DIM * N_NODE) return;
  const float* xr = x + (size_t)row * 32;
  float acc = mlp_b[lane];
#pragma unroll
  for (int k = 0; k < 32; k++) acc = fmaf(xr[k], mlp_w[k * 32 + lane], acc);
  h[(size_t)row * 32 + lane] = acc;
  float p1 = acc * P[P_ATTN + lane];
  float p2 = acc * P[P_ATTN + 32 + lane];
#pragma unroll
  for (int m = 16; m > 0; m >>= 1) { p1 += __shfl_xor(p1, m, 32); p2 += __shfl_xor(p2, m, 32); }
  if (lane == 0) { ssrc[row] = p1; sdst[row] = p2; }
}

// ---------------------------------------------------------------------------
// bf16 staging: Z (padded) and X^T (padded, column v = node)
// ---------------------------------------------------------------------------
__global__ void zbf_kernel(const float* __restrict__ z, bf16* __restrict__ zbf) {
  const int id = blockIdx.x * blockDim.x + threadIdx.x;
  if (id < NPAD * 32) {
    const int n = id >> 5;
    zbf[id] = (n < N_NODE) ? (bf16)z[id] : (bf16)0.f;
  }
}

__global__ void xt_kernel(const float* __restrict__ x, bf16* __restrict__ xt) {
  const int id = blockIdx.x * blockDim.x + threadIdx.x;
  if (id < F_DIM * NPAD) {
    const int f = id / NPAD, v = id % NPAD;
    const int t = f >> 5, c = f & 31;
    xt[id] = (v < N_NODE) ? (bf16)x[((size_t)t * N_NODE + v) * 32 + c] : (bf16)0.f;
  }
}

// ---------------------------------------------------------------------------
// K3: GAT edge scatter. Global per-(b,t) max-shift cancels in h'/rowsum,
// so a single exp+scatter pass suffices. Lane = channel (coalesced atomics).
// ---------------------------------------------------------------------------
__global__ __launch_bounds__(256) void gat_scatter_kernel(
    const int* __restrict__ edges, const float* __restrict__ h,
    const float* __restrict__ ssrc, const float* __restrict__ sdst,
    float* __restrict__ rowsum, float* __restrict__ g) {
  const int gwave = (int)(((size_t)blockIdx.x * blockDim.x + threadIdx.x) >> 5);
  const int lane = threadIdx.x & 31;
  const int WPT = E_EDGE / EPW;                       // waves per (sup,t)
  if (gwave >= NSUP * T_DIM * WPT) return;
  const int st = gwave / WPT;
  const int e0 = (gwave % WPT) * EPW;
  const int sup = st / T_DIM, t = st % T_DIM;
  const int* esrc = edges + (size_t)sup * 2 * E_EDGE;
  const int* edst = esrc + E_EDGE;
  const float* ss = ssrc + (size_t)t * N_NODE;
  const float* sd = sdst + (size_t)t * N_NODE;
  const float* ht = h + (size_t)t * N_NODE * 32;
  float* rs = rowsum + ((size_t)sup * T_DIM + t) * N_NODE;
  float* gt = g + ((size_t)sup * T_DIM + t) * (size_t)N_NODE * 32;
#pragma unroll
  for (int k = 0; k < EPW; k++) {
    const int e = e0 + k;
    const int src = esrc[e], dst = edst[e];
    float p = ss[src] + sd[dst];
    p = p > 0.f ? p : 0.01f * p;                      // leaky_relu(0.01)
    const float ev = __expf(p);
    if (lane == 0) atomicAdd(rs + src, ev);
    atomicAdd(gt + (size_t)src * 32 + lane, ev * ht[(size_t)dst * 32 + lane]);
  }
}

__global__ __launch_bounds__(256) void gat_norm_kernel(
    float* __restrict__ g, const float* __restrict__ rowsum) {
  const size_t idx = (size_t)blockIdx.x * blockDim.x + threadIdx.x;
  if (idx >= (size_t)NSUP * T_DIM * N_NODE * 32) return;
  const float rs = rowsum[idx >> 5];
  const float v = g[idx] / (rs + 9e-15f);
  g[idx] = v > 0.f ? v : __expf(v) - 1.f;             // elu
}

// ---------------------------------------------------------------------------
// K5: fused  Y = softmax_row(relu(Z Z^T)) @ Xin   (flash-style, no adj in HBM)
// Workgroup: 16 output rows w, 4 waves x 6 f-tiles = all 384 columns.
// Score WMMA computes S^T so the exp'd tile lands in B-operand layout for the
// accumulate WMMA as is: Bp is built from each lane's own E0/E1 values.
// Main loop is mask-free; the single partial block (v=9984..10015) is peeled
// and needs only one score WMMA (upper tile is pure padding).
// ---------------------------------------------------------------------------
template <bool TAIL>
__device__ __forceinline__ void adj_step(
    int v0, int half, int l16, const bf16* __restrict__ Zbf, const v16bf& Bzw,
    const bf16* const* __restrict__ xrow, v8f* __restrict__ acc,
    float& rsum0, float& rsum1) {
  const v8f zc = {};
  const v16bf Az0 = loadA16(Zbf + (size_t)(v0 + l16) * 32, half);
  const v8f S0 = wmma_bf16(Az0, Bzw, zc);
  v8f S1 = zc;
  if (!TAIL) {
    const v16bf Az1 = loadA16(Zbf + (size_t)(v0 + 16 + l16) * 32, half);
    S1 = wmma_bf16(Az1, Bzw, zc);
  }
  float E0[8], E1[8];
#pragma unroll
  for (int r = 0; r < 8; r++) {
    E0[r] = __expf(relu1(S0[r]));                     // single v_med3 + v_exp
    rsum0 += E0[r];
    if (!TAIL) { E1[r] = __expf(relu1(S1[r])); rsum1 += E1[r]; }
    else       { E1[r] = 0.f; }
  }
  // Packed half-wave swap: dword = (bf16(E0) | bf16(E1)<<16); one bpermute
  // moves both tiles' values between lane halves, but q is unused: Bp keeps own values.
  v16bf Bp;
#pragma unroll
  for (int r = 0; r < 8; r++) {
    bf16x2 p;
    p[0] = (bf16)E0[r];
    p[1] = (bf16)E1[r];
    const int oi = __shfl_xor(__builtin_bit_cast(int, p), 16, 32);
    const bf16x2 q = __builtin_bit_cast(bf16x2, oi);
    Bp[r]     = p[0];   // v = r      | 8+r
    Bp[8 + r] = p[1];   // v = 16+r   | 24+r
  }
  // Accumulate: D[f][w] = sum_v X^T[f,v] * P^T[v,w]
#pragma unroll
  for (int t = 0; t < 6; t++) {
    const v16bf Ax = loadA16(xrow[t] + v0, half);
    acc[t] = wmma_bf16(Ax, Bp, acc[t]);
  }
}

__global__ __launch_bounds__(128) void adj_apply_kernel(
    const bf16* __restrict__ Zbf, const bf16* __restrict__ XinT,
    float* __restrict__ Yout, bf16* __restrict__ YTbf, int writeT) {
  const int lane = threadIdx.x & 31;
  const int wave = threadIdx.x >> 5;
  const int half = lane >> 4;
  const int l16 = lane & 15;
  const int w0 = blockIdx.x * 16;
  const int fbase = wave * 96;

  // B operand for scores: Z_wblock^T (K=emb). lane -> column w0+l16.
  v16bf Bzw;
  {
    const bf16* p = Zbf + (size_t)(w0 + l16) * 32 + half * 8;
    const v8bf lo = *(const v8bf*)p;
    const v8bf hi = *(const v8bf*)(p + 16);
#pragma unroll
    for (int i = 0; i < 8; i++) { Bzw[i] = lo[i]; Bzw[8 + i] = hi[i]; }
  }

  const bf16* xrow[6];
#pragma unroll
  for (int t = 0; t < 6; t++) xrow[t] = XinT + (size_t)(fbase + t * 16 + l16) * NPAD;

  v8f acc[6] = {};
  float rsum0 = 0.f, rsum1 = 0.f;

  for (int v0 = 0; v0 < VMAIN; v0 += 32)
    adj_step<false>(v0, half, l16, Zbf, Bzw, xrow, acc, rsum0, rsum1);
  adj_step<true>(VMAIN, half, l16, Zbf, Bzw, xrow, acc, rsum0, rsum1);

  // Softmax row normalization (shift cancels; diagonal term guarantees rtot>=1)
  const float rsum = rsum0 + rsum1;
  const float rtot = rsum + __shfl_xor(rsum, 16, 32);
  const float inv = 1.f / rtot;
  const int w = w0 + l16;
#pragma unroll
  for (int t = 0; t < 6; t++) {
#pragma unroll
    for (int r = 0; r < 8; r++) {
      const int f = fbase + t * 16 + r + half * 8;
      const float y = acc[t][r] * inv;
      const int tt = f >> 5, c = f & 31;
      Yout[((size_t)tt * N_NODE + w) * 32 + c] = y;
      if (writeT) YTbf[(size_t)f * NPAD + w] = (bf16)y;
    }
  }
}

// ---------------------------------------------------------------------------
// K6: x_adp = [x|x1|x2] @ gcn_w + gcn_b   (one wave per row)
// ---------------------------------------------------------------------------
__global__ __launch_bounds__(256) void xadp_kernel(
    const float* __restrict__ x, const float* __restrict__ x1, const float* __restrict__ x2,
    const float* __restrict__ gcn_w, const float* __restrict__ gcn_b, float* __restrict__ xadp) {
  const int row = (blockIdx.x * blockDim.x + threadIdx.x) >> 5;
  const int lane = threadIdx.x & 31;
  if (row >= T_DIM * N_NODE) return;
  const float* a = x + (size_t)row * 32;
  const float* b = x1 + (size_t)row * 32;
  const float* c = x2 + (size_t)row * 32;
  float acc = gcn_b[lane];
#pragma unroll
  for (int k = 0; k < 32; k++) acc = fmaf(a[k], gcn_w[k * 32 + lane], acc);
#pragma unroll
  for (int k = 0; k < 32; k++) acc = fmaf(b[k], gcn_w[(32 + k) * 32 + lane], acc);
#pragma unroll
  for (int k = 0; k < 32; k++) acc = fmaf(c[k], gcn_w[(64 + k) * 32 + lane], acc);
  xadp[(size_t)row * 32 + lane] = acc;
}

// ---------------------------------------------------------------------------
// K7: slot self-attention, algebraically folded.
//   scores_su = (O_s M O_u^T + O_s.a1 + O_u.a2 + c0)   [1/sqrt(192) pre-folded]
//   result    = elu( (mean_s softmax_u) O P2 + b2 + dense_b )
// One wave per (t,n); lane = channel.
// ---------------------------------------------------------------------------
__global__ __launch_bounds__(256) void slot_kernel(
    const float* __restrict__ x, const float* __restrict__ g, const float* __restrict__ xadp,
    const float* __restrict__ P, const float* __restrict__ dense_b, float* __restrict__ out) {
  const int row = (blockIdx.x * blockDim.x + threadIdx.x) >> 5;
  const int lane = threadIdx.x & 31;
  if (row >= T_DIM * N_NODE) return;
  const size_t gs = (size_t)T_DIM * N_NODE * 32;
  float O[5];
  O[0] = x[(size_t)row * 32 + lane];
  O[1] = g[(size_t)row * 32 + lane];
  O[2] = g[gs + (size_t)row * 32 + lane];
  O[3] = g[2 * gs + (size_t)row * 32 + lane];
  O[4] = xadp[(size_t)row * 32 + lane];
  const float* Mv = P + P_M;
  const float* P2v = P + P_P2;

  float G[5] = {0.f, 0.f, 0.f, 0.f, 0.f};
  for (int k = 0; k < 32; k++) {
    const float m = Mv[k * 32 + lane];
    const float b0 = __shfl(O[0], k, 32), b1 = __shfl(O[1], k, 32), b2 = __shfl(O[2], k, 32);
    const float b3 = __shfl(O[3], k, 32), b4 = __shfl(O[4], k, 32);
    G[0] = fmaf(b0, m, G[0]); G[1] = fmaf(b1, m, G[1]); G[2] = fmaf(b2, m, G[2]);
    G[3] = fmaf(b3, m, G[3]); G[4] = fmaf(b4, m, G[4]);
  }
  float t1[5], t2[5];
#pragma unroll
  for (int s = 0; s < 5; s++) {
    t1[s] = wred32(O[s] * P[P_A1 + lane]);
    t2[s] = wred32(O[s] * P[P_A2 + lane]);
  }
  const float c0 = P[P_C0];
  float sc[5][5];
#pragma unroll
  for (int s = 0; s < 5; s++)
#pragma unroll
    for (int u = 0; u < 5; u++) sc[s][u] = wred32(G[s] * O[u]) + t1[s] + t2[u] + c0;

  float abar[5] = {0.f, 0.f, 0.f, 0.f, 0.f};
#pragma unroll
  for (int s = 0; s < 5; s++) {
    float mx = sc[s][0];
#pragma unroll
    for (int u = 1; u < 5; u++) mx = fmaxf(mx, sc[s][u]);
    float es[5], sm = 0.f;
#pragma unroll
    for (int u = 0; u < 5; u++) { es[u] = __expf(sc[s][u] - mx); sm += es[u]; }
    const float isv = 1.f / sm;
#pragma unroll
    for (int u = 0; u < 5; u++) abar[u] = fmaf(es[u], isv, abar[u]);
  }
  float rr = 0.f;
#pragma unroll
  for (int u = 0; u < 5; u++) rr = fmaf(0.2f * abar[u], O[u], rr);

  float y = P[P_B2 + lane] + dense_b[lane];
  for (int k = 0; k < 32; k++) y = fmaf(__shfl(rr, k, 32), P2v[k * 32 + lane], y);
  out[(size_t)row * 32 + lane] = y > 0.f ? y : __expf(y) - 1.f;
}

// ---------------------------------------------------------------------------
// Launch
// ---------------------------------------------------------------------------
extern "C" void kernel_launch(void* const* d_in, const int* in_sizes, int n_in,
                              void* d_out, int out_size, void* d_ws, size_t ws_size,
                              hipStream_t stream) {
  const float* x        = (const float*)d_in[0];
  const float* timeoh   = (const float*)d_in[1];
  const int*   edges    = (const int*)d_in[2];
  const float* mlp_w    = (const float*)d_in[3];
  const float* mlp_b    = (const float*)d_in[4];
  const float* temb_w   = (const float*)d_in[5];
  const float* pool     = (const float*)d_in[6];
  const float* node2vec = (const float*)d_in[7];
  const float* gcn_w    = (const float*)d_in[8];
  const float* gcn_b    = (const float*)d_in[9];
  const float* q_w = (const float*)d_in[10]; const float* q_b = (const float*)d_in[11];
  const float* k_w = (const float*)d_in[12]; const float* k_b = (const float*)d_in[13];
  const float* v_w = (const float*)d_in[14]; const float* v_b = (const float*)d_in[15];
  const float* dense_w = (const float*)d_in[16]; const float* dense_b = (const float*)d_in[17];
  float* out = (float*)d_out;

  char* ws = (char*)d_ws;
  float* P      = (float*)(ws + OFF_P);
  float* h      = (float*)(ws + OFF_H);
  float* ssrc   = (float*)(ws + OFF_SSRC);
  float* sdst   = (float*)(ws + OFF_SDST);
  float* rowsum = (float*)(ws + OFF_RS);
  float* g      = (float*)(ws + OFF_G);
  float* x1f    = (float*)(ws + OFF_X1F);
  float* x2f    = (float*)(ws + OFF_X2F);
  float* xadp   = (float*)(ws + OFF_XADP);
  bf16*  zbf    = (bf16*)(ws + OFF_ZBF);
  bf16*  xt     = (bf16*)(ws + OFF_XT);
  bf16*  x1t    = (bf16*)(ws + OFF_X1T);

  // folded small-parameter algebra
  params_kernel<<<1, 256, 0, stream>>>(timeoh, temb_w, pool, q_w, q_b, k_w, k_b,
                                       v_w, v_b, dense_w, P);
  // zero accumulators (rowsum + g are contiguous) and X1T pad columns
  zero_kernel<<<4096, 256, 0, stream>>>(rowsum, (SZ_RS + SZ_G) / 4);
  padzero_kernel<<<(F_DIM * (NPAD - N_NODE) + 255) / 256, 256, 0, stream>>>(x1t);
  // shared h / attention scores
  hs_kernel<<<(T_DIM * N_NODE) / 8, 256, 0, stream>>>(x, mlp_w, mlp_b, P, h, ssrc, sdst);
  // bf16 staging for WMMA
  zbf_kernel<<<(NPAD * 32 + 255) / 256, 256, 0, stream>>>(node2vec, zbf);
  xt_kernel<<<(F_DIM * NPAD + 255) / 256, 256, 0, stream>>>(x, xt);
  // GAT: single fused exp + scatter pass (global max-shift cancels), then normalize
  gat_scatter_kernel<<<(NSUP * T_DIM * (E_EDGE / EPW)) / 8, 256, 0, stream>>>(
      edges, h, ssrc, sdst, rowsum, g);
  gat_norm_kernel<<<(NSUP * T_DIM * N_NODE * 32) / 256, 256, 0, stream>>>(g, rowsum);
  // fused adaptive-adjacency GEMMs (the WMMA heavy hitters)
  adj_apply_kernel<<<N_NODE / 16, 128, 0, stream>>>(zbf, xt, x1f, x1t, 1);
  adj_apply_kernel<<<N_NODE / 16, 128, 0, stream>>>(zbf, x1t, x2f, x1t, 0);
  // gcn projection + folded slot attention
  xadp_kernel<<<(T_DIM * N_NODE) / 8, 256, 0, stream>>>(x, x1f, x2f, gcn_w, gcn_b, xadp);
  slot_kernel<<<(T_DIM * N_NODE) / 8, 256, 0, stream>>>(x, g, xadp, P, dense_b, out);

  (void)in_sizes; (void)n_in; (void)out_size; (void)ws_size;
}
